// SwitchableQATGPT2MLP_65429531787889
// MI455X (gfx1250) — matrix operand order repaired and checked by reading, not compiled, not run on hardware
//
#include <hip/hip_runtime.h>
#include <hip/hip_bf16.h>
#include <math.h>

typedef __attribute__((ext_vector_type(16))) __bf16 bf16x16;
typedef __attribute__((ext_vector_type(8)))  __bf16 bf16x8;
typedef __attribute__((ext_vector_type(8)))  float  f32x8;

// ---------------------------------------------------------------------------
// CDNA5 async global->LDS copy (ASYNCcnt-tracked), per-lane 16B
// ---------------------------------------------------------------------------
__device__ __forceinline__ void async_ld_b128(unsigned lds_off, const void* gptr) {
  asm volatile("global_load_async_to_lds_b128 %0, %1, off"
               :: "v"(lds_off), "v"((unsigned long long)(uintptr_t)gptr)
               : "memory");
}
__device__ __forceinline__ void wait_async0() {
  asm volatile("s_wait_asynccnt 0" ::: "memory");
}

// branch-free erf (Abramowitz-Stegun 7.1.26, |err| < 1.5e-7)
__device__ __forceinline__ float fast_erf(float x) {
  float ax = fabsf(x);
  float t  = __builtin_amdgcn_rcpf(__builtin_fmaf(0.3275911f, ax, 1.0f));
  float p  = __builtin_fmaf(t, 1.061405429f, -1.453152027f);
  p = __builtin_fmaf(t, p, 1.421413741f);
  p = __builtin_fmaf(t, p, -0.284496736f);
  p = __builtin_fmaf(t, p, 0.254829592f);
  p = p * t;
  float e = __expf(-ax * ax);               // v_exp_f32 path
  float r = __builtin_fmaf(-p, e, 1.0f);
  return copysignf(r, x);
}

// ---------------------------------------------------------------------------
// Small utility kernels
// ---------------------------------------------------------------------------

__global__ void zero_scales_kernel(unsigned* s) {
  if (threadIdx.x < 2) s[threadIdx.x] = 0u;
}

// absmax via bit-pattern max (valid for non-negative floats)
__global__ void absmax_kernel(const float* __restrict__ w, int n, unsigned* out) {
  __shared__ unsigned red[256];
  float m = 0.f;
  for (int i = blockIdx.x * blockDim.x + threadIdx.x; i < n; i += gridDim.x * blockDim.x)
    m = fmaxf(m, fabsf(w[i]));
  red[threadIdx.x] = __float_as_uint(m);
  __syncthreads();
  for (int s = 128; s > 0; s >>= 1) {
    if ((int)threadIdx.x < s) {
      unsigned a = red[threadIdx.x], b = red[threadIdx.x + s];
      red[threadIdx.x] = (a > b) ? a : b;
    }
    __syncthreads();
  }
  if (threadIdx.x == 0) atomicMax(out, red[0]);
}

// Pack W (fake-quantized, bf16) and lora B (scaled) into [N, Kc] cat buffer.
__global__ void quant_pack_kernel(const float* __restrict__ W,
                                  const float* __restrict__ Bl,
                                  const unsigned* __restrict__ maxbits,
                                  __bf16* __restrict__ out,
                                  int K, int Kc, float lscale) {
  int n = blockIdx.y;
  int k = blockIdx.x * blockDim.x + threadIdx.x;
  if (k >= Kc) return;
  float v;
  if (k < K) {
    float mx    = __uint_as_float(maxbits[0]);
    float scale = mx * (1.0f / 127.0f);
    float q = rintf(W[(size_t)n * K + k] * (127.0f / mx)); // RTE == jnp.round
    q = fminf(fmaxf(q, -128.0f), 127.0f);
    v = q * scale;
  } else if (k < K + 16) {
    v = lscale * Bl[n * 16 + (k - K)];
  } else {
    v = 0.0f;
  }
  out[(size_t)n * Kc + k] = (__bf16)v;
}

// fp32 activations -> bf16 cat buffer cols [0,K)
__global__ void convert_x_kernel(const float* __restrict__ X,
                                 __bf16* __restrict__ out, int K, int Kc) {
  int m = blockIdx.y;
  int k = blockIdx.x * blockDim.x + threadIdx.x;
  if (k < K) out[(size_t)m * Kc + k] = (__bf16)X[(size_t)m * K + k];
}

// T[m][r] = sum_k Xc[m][k] * A[r][k] for r<16, 0 for r in [16,32);
// written into the SAME cat buffer at columns [K, K+32).
__global__ void lora_t_kernel(const __bf16* Xc, int ldx,
                              const float* __restrict__ A, int K,
                              __bf16* outc, int ldo) {
  int t = blockIdx.x * blockDim.x + threadIdx.x;
  int m = t >> 5;
  int r = t & 31;
  float s = 0.f;
  if (r < 16) {
    const __bf16* x = Xc + (size_t)m * ldx;
    const float*  a = A + (size_t)r * K;
    for (int k = 0; k < K; ++k) s += (float)x[k] * a[k];
  }
  outc[(size_t)m * ldo + K + r] = (__bf16)s;
}

// ---------------------------------------------------------------------------
// Main WMMA GEMM:  C[M,N] = act( Acat[M,Kc] * Bcat[N,Kc]^T + bias )
// Block tile 128x128, BK=32, 8 waves in 2(M)x4(N), wave tile 64x32 (4x2 wmma).
// Global->LDS via CDNA5 async copies, double buffered.
// EPI==0: exact-erf GELU (A&S approx), write bf16; EPI==1: identity, fp32.
// ---------------------------------------------------------------------------
#define LSTR 40               // LDS row stride in halves (32 data + 8 pad)
#define TILEH (128 * LSTR)    // halves per tile buffer

template <int EPI>
__global__ __launch_bounds__(256)
void gemm_qat_kernel(const __bf16* __restrict__ Am, int lda,
                     const __bf16* __restrict__ Bm, int ldb,
                     const float* __restrict__ bias,
                     void* __restrict__ Cout, int ldc,
                     int Kc) {
  __shared__ __bf16 sA[2 * TILEH];
  __shared__ __bf16 sB[2 * TILEH];

  const int tid  = threadIdx.x;
  const int lane = tid & 31;
  const int wave = tid >> 5;
  const int wm   = wave >> 2;        // 0..1  -> M offset wm*64
  const int wn   = wave & 3;         // 0..3  -> N offset wn*32
  const int bm   = blockIdx.y * 128;
  const int bn   = blockIdx.x * 128;

  // loader mapping: each thread async-copies 16 halves (two 16B chunks) per tile
  const int ldr_row = tid >> 1;            // 0..127
  const int ldr_col = (tid & 1) * 16;      // 0 or 16 (halves)

  // LDS byte offsets (flat shared ptr low 32 bits == LDS offset)
  const unsigned sA0 = (unsigned)(uintptr_t)(void*)&sA[ldr_row * LSTR + ldr_col];
  const unsigned sB0 = (unsigned)(uintptr_t)(void*)&sB[ldr_row * LSTR + ldr_col];
  const __bf16* gA0 = Am + (size_t)(bm + ldr_row) * lda + ldr_col;
  const __bf16* gB0 = Bm + (size_t)(bn + ldr_row) * ldb + ldr_col;

  // fragment addressing (ISA 7.12.2)
  const int frow = lane & 15;
  const int kbA  = (lane >> 4) * 8;        // A: K-blocks {0..7|16..23} / {8..15|24..31}
  const int kbB  = (lane >> 4) * 8;        // B: K-blocks {0..7|16..23} / {8..15|24..31}

  f32x8 acc[4][2] = {};
  const int KT = Kc / 32;

  // prologue: async-copy tile 0 into buffer 0
  async_ld_b128(sA0,                  gA0);
  async_ld_b128(sA0 + 16,             gA0 + 8);
  async_ld_b128(sB0,                  gB0);
  async_ld_b128(sB0 + 16,             gB0 + 8);
  wait_async0();
  __syncthreads();

  for (int kt = 0; kt < KT; ++kt) {
    const int buf = kt & 1;
    if (kt + 1 < KT) {
      // issue async copies for next tile into the other buffer (no wait yet)
      const unsigned nb = (buf ^ 1) * (TILEH * 2);   // bytes
      const __bf16* gA = gA0 + (kt + 1) * 32;
      const __bf16* gB = gB0 + (kt + 1) * 32;
      async_ld_b128(sA0 + nb,      gA);
      async_ld_b128(sA0 + nb + 16, gA + 8);
      async_ld_b128(sB0 + nb,      gB);
      async_ld_b128(sB0 + nb + 16, gB + 8);
    }

    // compute on current buffer
    bf16x16 af[4];
    bf16x16 bfr[2];
#pragma unroll
    for (int mt = 0; mt < 4; ++mt) {
      const __bf16* p = &sA[buf * TILEH + (wm * 64 + mt * 16 + frow) * LSTR + kbA];
      bf16x8 lo = *(const bf16x8*)p;
      bf16x8 hi = *(const bf16x8*)(p + 16);
      af[mt] = __builtin_shufflevector(lo, hi, 0, 1, 2, 3, 4, 5, 6, 7,
                                       8, 9, 10, 11, 12, 13, 14, 15);
    }
#pragma unroll
    for (int nt = 0; nt < 2; ++nt) {
      const __bf16* p = &sB[buf * TILEH + (wn * 32 + nt * 16 + frow) * LSTR + kbB];
      bf16x8 lo = *(const bf16x8*)p;
      bf16x8 hi = *(const bf16x8*)(p + 16);
      bfr[nt] = __builtin_shufflevector(lo, hi, 0, 1, 2, 3, 4, 5, 6, 7,
                                        8, 9, 10, 11, 12, 13, 14, 15);
    }
#pragma unroll
    for (int mt = 0; mt < 4; ++mt)
#pragma unroll
      for (int nt = 0; nt < 2; ++nt)
        acc[mt][nt] = __builtin_amdgcn_wmma_f32_16x16x32_bf16(
            false, af[mt], false, bfr[nt], (short)0, acc[mt][nt], false, false);

    if (kt + 1 < KT) wait_async0();   // async writes for next buffer complete
    __syncthreads();                  // visible to all waves; current buffer free
  }

  // epilogue: C/D layout -> lane owns col n=lane%16, rows mh..mh+7
  const int nl = lane & 15;
  const int mh = (lane >> 4) * 8;
#pragma unroll
  for (int mt = 0; mt < 4; ++mt) {
#pragma unroll
    for (int nt = 0; nt < 2; ++nt) {
      const int gn = bn + wn * 32 + nt * 16 + nl;
      const float bv = bias[gn];
      const size_t gm0 = (size_t)bm + wm * 64 + mt * 16 + mh;
#pragma unroll
      for (int v = 0; v < 8; ++v) {
        float val = acc[mt][nt][v] + bv;
        if (EPI == 0) {
          float g = 0.5f * val * (1.0f + fast_erf(val * 0.70710678118654752f));
          ((__bf16*)Cout)[(gm0 + v) * (size_t)ldc + gn] = (__bf16)g;
        } else {
          ((float*)Cout)[(gm0 + v) * (size_t)ldc + gn] = val;
        }
      }
    }
  }
}

// ---------------------------------------------------------------------------
// Launch
// ---------------------------------------------------------------------------
extern "C" void kernel_launch(void* const* d_in, const int* in_sizes, int n_in,
                              void* d_out, int out_size, void* d_ws, size_t ws_size,
                              hipStream_t stream) {
  const float* X      = (const float*)d_in[0];  // [4,2048,1024]
  const float* W_fc   = (const float*)d_in[1];  // [4096,1024]
  const float* b_fc   = (const float*)d_in[2];  // [4096]
  const float* A_fc   = (const float*)d_in[3];  // [16,1024]
  const float* B_fc   = (const float*)d_in[4];  // [4096,16]
  const float* W_pj   = (const float*)d_in[5];  // [1024,4096]
  const float* b_pj   = (const float*)d_in[6];  // [1024]
  const float* A_pj   = (const float*)d_in[7];  // [16,4096]
  const float* B_pj   = (const float*)d_in[8];  // [1024,16]
  float* out = (float*)d_out;                   // [8192,1024]

  const int M = 8192, D = 1024, F = 4096;
  const int Kc1 = 1056;   // 1024 + 16 lora + 16 pad  (33 * 32)
  const int Kc2 = 4128;   // 4096 + 16 lora + 16 pad  (129 * 32)
  const float SCALING = 2.0f;  // lora_alpha / lora_rank

  char* ws = (char*)d_ws;
  size_t off = 0;
  unsigned* scales = (unsigned*)(ws + off); off += 256;
  __bf16* Xcat = (__bf16*)(ws + off); off += (size_t)M * Kc1 * 2;   // 16.5 MB
  __bf16* Wfc  = (__bf16*)(ws + off); off += (size_t)F * Kc1 * 2;   //  8.3 MB
  __bf16* Wpj  = (__bf16*)(ws + off); off += (size_t)D * Kc2 * 2;   //  8.1 MB
  __bf16* Hcat = (__bf16*)(ws + off); off += (size_t)M * Kc2 * 2;   // 64.5 MB
  (void)ws_size; (void)in_sizes; (void)n_in; (void)out_size;

  // 1) fake-quant scales (re-zero every call: ws is not re-poisoned)
  zero_scales_kernel<<<1, 64, 0, stream>>>(scales);
  absmax_kernel<<<1024, 256, 0, stream>>>(W_fc, F * D, scales + 0);
  absmax_kernel<<<1024, 256, 0, stream>>>(W_pj, D * F, scales + 1);

  // 2) pack weights: [Wq | SCALING*B | 0] as bf16
  quant_pack_kernel<<<dim3((Kc1 + 255) / 256, F), 256, 0, stream>>>(
      W_fc, B_fc, scales + 0, Wfc, D, Kc1, SCALING);
  quant_pack_kernel<<<dim3((Kc2 + 255) / 256, D), 256, 0, stream>>>(
      W_pj, B_pj, scales + 1, Wpj, F, Kc2, SCALING);

  // 3) activations: x -> bf16 cat buffer, then lora T1 columns
  convert_x_kernel<<<dim3(D / 256, M), 256, 0, stream>>>(X, Xcat, D, Kc1);
  lora_t_kernel<<<(M * 32) / 256, 256, 0, stream>>>(Xcat, Kc1, A_fc, D, Xcat, Kc1);

  // 4) GEMM1 + bias + GELU -> Hcat (bf16)
  gemm_qat_kernel<0><<<dim3(F / 128, M / 128), 256, 0, stream>>>(
      Xcat, Kc1, Wfc, Kc1, b_fc, (void*)Hcat, Kc2, Kc1);

  // 5) lora T2 columns of Hcat
  lora_t_kernel<<<(M * 32) / 256, 256, 0, stream>>>(Hcat, Kc2, A_pj, F, Hcat, Kc2);

  // 6) GEMM2 + bias -> fp32 output
  gemm_qat_kernel<1><<<dim3(D / 128, M / 128), 256, 0, stream>>>(
      Hcat, Kc2, Wpj, Kc2, b_pj, (void*)out, D, Kc2);
}
